// LSTMAggregator_1382979469563
// MI455X (gfx1250) — compile-verified, matrix operand order checked by reading, not run on hardware
//
#include <hip/hip_runtime.h>

#define IN_DIM  128
#define UNITS   128
#define BATCH   16384
#define N_NEIGH 25
#define TILES   (BATCH / 16)   // 1024 batch tiles of 16 rows
#define GRID    256            // persistent-ish: each WG handles TILES/GRID tiles

typedef __attribute__((ext_vector_type(16))) __bf16 bf16x16;
typedef __attribute__((ext_vector_type(8)))  float  f32x8;

union Frag { unsigned int u[8]; bf16x16 v; };

static __device__ __forceinline__ unsigned int bf_rn(float f) {
    return (__float_as_uint(f) + 0x8000u) >> 16;   // round-to-nearest bf16 (cheap)
}
static __device__ __forceinline__ unsigned int pack2(float a, float b) {
    return bf_rn(a) | ((__float_as_uint(b) + 0x8000u) & 0xffff0000u);
}
static __device__ __forceinline__ f32x8 splat8(float x) {
    f32x8 r;
#pragma unroll
    for (int i = 0; i < 8; ++i) r[i] = x;
    return r;
}
static __device__ __forceinline__ float sigf(float x) {
    return 1.0f / (1.0f + __expf(-x));
}
static __device__ __forceinline__ float tanh_fast(float x) {
    return 2.0f / (1.0f + __expf(-2.0f * x)) - 1.0f;   // overflow-safe
}
static __device__ __forceinline__ int koff(int v, int s0) {
    return ((v < 4) ? 2 * v : 16 + 2 * (v - 4)) + s0;
}

// Read 4 A-fragments (16x32 bf16 each, K=128) from a row-major [16][128] bf16 LDS buffer.
static __device__ __forceinline__ void ld_afrag(Frag* A, const char* base, int m, int s0) {
#pragma unroll
    for (int a = 0; a < 4; ++a) {
        const char* p = base + m * 256 + (32 * a + s0) * 2;
        uint4 lo = *(const uint4*)p;
        uint4 hi = *(const uint4*)(p + 32);
        A[a].u[0] = lo.x; A[a].u[1] = lo.y; A[a].u[2] = lo.z; A[a].u[3] = lo.w;
        A[a].u[4] = hi.x; A[a].u[5] = hi.y; A[a].u[6] = hi.z; A[a].u[7] = hi.w;
    }
}

__global__ __launch_bounds__(256) void lstm_agg_kernel(
    const float* __restrict__ features,
    const int*   __restrict__ node,
    const int*   __restrict__ neighbours,
    const float* __restrict__ node_weights,
    const float* __restrict__ lstm_W,
    const float* __restrict__ lstm_U,
    const float* __restrict__ lstm_b,
    float*       __restrict__ out)
{
    __shared__ char smem[16384];
    char* xbuf0 = smem;            // 4 KB : x_t  staging  (bf16 [16][128])
    char* xbuf1 = smem + 4096;     // 4 KB : x_t+1 staging
    char* hbuf0 = smem + 8192;     // 4 KB : h ping
    char* hbuf1 = smem + 12288;    // 4 KB : h pong
    char* xb[2] = { xbuf0, xbuf1 };
    char* hb[2] = { hbuf0, hbuf1 };

    const int tid  = threadIdx.x;
    const int lane = tid & 31;
    const int wave = tid >> 5;           // hidden-slice owner: units [16*wave, 16*wave+16)
    const int m    = lane & 15;
    const int s0   = (lane & 16) ? 8 : 0;
    const int col0 = wave * 16 + m;      // this lane's output column within UNITS

    // ---- Register-resident B fragments: W, U (4 gates x 4 K-blocks) + node_weights ----
    Frag BW[4][4], BU[4][4], BN[4];
#pragma unroll
    for (int g = 0; g < 4; ++g) {
        const float* wsrc = lstm_W + (g * 128 + col0);
        const float* usrc = lstm_U + (g * 128 + col0);
#pragma unroll
        for (int a = 0; a < 4; ++a)
#pragma unroll
            for (int v = 0; v < 8; ++v) {
                int k = 32 * a + koff(v, s0);
                BW[g][a].u[v] = pack2(wsrc[(size_t)k * 512], wsrc[(size_t)(k + 1) * 512]);
                BU[g][a].u[v] = pack2(usrc[(size_t)k * 512], usrc[(size_t)(k + 1) * 512]);
            }
    }
#pragma unroll
    for (int a = 0; a < 4; ++a)
#pragma unroll
        for (int v = 0; v < 8; ++v) {
            int k = 32 * a + koff(v, s0);
            BN[a].u[v] = pack2(node_weights[k * 128 + col0], node_weights[(k + 1) * 128 + col0]);
        }
    float bias[4];
#pragma unroll
    for (int g = 0; g < 4; ++g) bias[g] = lstm_b[g * 128 + col0];

    // Cooperative x-staging role: 16 threads per row, 8 floats each.
    const int str = tid >> 4;            // staging row 0..15
    const int stc = (tid & 15) * 8;      // staging float column base

    for (int tile = blockIdx.x; tile < TILES; tile += GRID) {
        const int baseRow = tile * 16;

        // Stage x for t = 0.
        {
            int nidx = neighbours[(baseRow + str) * N_NEIGH + 0];
            const float* rp = features + (size_t)nidx * IN_DIM + stc;
            float4 f0 = *(const float4*)rp;
            float4 f1 = *(const float4*)(rp + 4);
            uint4 w;
            w.x = pack2(f0.x, f0.y); w.y = pack2(f0.z, f0.w);
            w.z = pack2(f1.x, f1.y); w.w = pack2(f1.z, f1.w);
            *(uint4*)(xb[0] + str * 256 + stc * 2) = w;
        }
        __syncthreads();

        f32x8 c    = splat8(0.0f);
        f32x8 hfin = splat8(0.0f);
        Frag  hA[4];
#pragma unroll
        for (int a = 0; a < 4; ++a)
#pragma unroll
            for (int v = 0; v < 8; ++v) hA[a].u[v] = 0u;   // h0 = 0

        for (int t = 0; t < N_NEIGH; ++t) {
            Frag xA[4];
            ld_afrag(xA, xb[t & 1], m, s0);
            if (t > 0) ld_afrag(hA, hb[t & 1], m, s0);

            // Stage x for t+1 (and prefetch t+2 rows).
            if (t + 1 < N_NEIGH) {
                int nidx = neighbours[(baseRow + str) * N_NEIGH + (t + 1)];
                if (t + 2 < N_NEIGH) {
                    int nn = neighbours[(baseRow + str) * N_NEIGH + (t + 2)];
                    __builtin_prefetch(features + (size_t)nn * IN_DIM + stc, 0, 1);
                }
                const float* rp = features + (size_t)nidx * IN_DIM + stc;
                float4 f0 = *(const float4*)rp;
                float4 f1 = *(const float4*)(rp + 4);
                uint4 w;
                w.x = pack2(f0.x, f0.y); w.y = pack2(f0.z, f0.w);
                w.z = pack2(f1.x, f1.y); w.w = pack2(f1.z, f1.w);
                *(uint4*)(xb[(t + 1) & 1] + str * 256 + stc * 2) = w;
            }

            // z = b + x_t @ W + h_{t-1} @ U : 32 register-operand WMMAs, no LDS in the chain.
            f32x8 z[4];
#pragma unroll
            for (int g = 0; g < 4; ++g) {
                f32x8 acc = splat8(bias[g]);
#pragma unroll
                for (int a = 0; a < 4; ++a)
                    acc = __builtin_amdgcn_wmma_f32_16x16x32_bf16(
                        false, xA[a].v, false, BW[g][a].v, (short)0, acc, false, false);
#pragma unroll
                for (int a = 0; a < 4; ++a)
                    acc = __builtin_amdgcn_wmma_f32_16x16x32_bf16(
                        false, hA[a].v, false, BU[g][a].v, (short)0, acc, false, false);
                z[g] = acc;
            }

            // Gate nonlinearities + state update (TRANS ops co-execute with WMMA).
            char* hw = hb[(t + 1) & 1];
#pragma unroll
            for (int j = 0; j < 8; ++j) {
                float iv = sigf(z[0][j]);
                float fv = sigf(z[1][j]);
                float gv = tanh_fast(z[2][j]);
                float ov = sigf(z[3][j]);
                float cj = fv * c[j] + iv * gv;
                c[j] = cj;
                float hj = ov * tanh_fast(cj);
                hfin[j] = hj;
                *(unsigned short*)(hw + (j + s0) * 256 + col0 * 2) = (unsigned short)bf_rn(hj);
            }
            __syncthreads();
        }

        // from_self = features[node] @ node_weights ; out = relu((self + h_T)/2)
        {
            int sidx = node[baseRow + str];
            const float* rp = features + (size_t)sidx * IN_DIM + stc;
            float4 f0 = *(const float4*)rp;
            float4 f1 = *(const float4*)(rp + 4);
            uint4 w;
            w.x = pack2(f0.x, f0.y); w.y = pack2(f0.z, f0.w);
            w.z = pack2(f1.x, f1.y); w.w = pack2(f1.z, f1.w);
            *(uint4*)(xb[0] + str * 256 + stc * 2) = w;
        }
        __syncthreads();

        Frag sA[4];
        ld_afrag(sA, xb[0], m, s0);
        f32x8 acc = splat8(0.0f);
#pragma unroll
        for (int a = 0; a < 4; ++a)
            acc = __builtin_amdgcn_wmma_f32_16x16x32_bf16(
                false, sA[a].v, false, BN[a].v, (short)0, acc, false, false);
#pragma unroll
        for (int j = 0; j < 8; ++j) {
            float o = (acc[j] + hfin[j]) * 0.5f;
            o = o > 0.0f ? o : 0.0f;
            out[(size_t)(baseRow + j + s0) * UNITS + col0] = o;
        }
        __syncthreads();   // protect xbuf/hbuf reuse by the next tile
    }
}

extern "C" void kernel_launch(void* const* d_in, const int* in_sizes, int n_in,
                              void* d_out, int out_size, void* d_ws, size_t ws_size,
                              hipStream_t stream) {
    (void)in_sizes; (void)n_in; (void)d_ws; (void)ws_size; (void)out_size;
    const float* features     = (const float*)d_in[0];
    const int*   node         = (const int*)d_in[1];
    const int*   neighbours   = (const int*)d_in[2];
    const float* node_weights = (const float*)d_in[3];
    const float* lstm_W       = (const float*)d_in[4];
    const float* lstm_U       = (const float*)d_in[5];
    const float* lstm_b       = (const float*)d_in[6];
    float* out = (float*)d_out;

    lstm_agg_kernel<<<dim3(GRID), dim3(256), 0, stream>>>(
        features, node, neighbours, node_weights, lstm_W, lstm_U, lstm_b, out);
}
